// MultitaskLSTMAttnClassifiers_29016799051921
// MI455X (gfx1250) — matrix operand order repaired and checked by reading, not compiled, not run on hardware
//
#include <hip/hip_runtime.h>
#include <math.h>
#include <stdint.h>

// Problem constants (from reference)
#define BB   64      // batch
#define CC   1024    // channels / LSTM input
#define TT   16      // time steps
#define HHID 1024    // hidden
#define G4   4096    // 4*H
#define NTASK 2
#define NC0  125
#define NC1  352

typedef _Float16 v16h __attribute__((ext_vector_type(16)));
typedef float    v8f  __attribute__((ext_vector_type(8)));
typedef int          v4i   __attribute__((ext_vector_type(4)));
typedef unsigned int u32x4 __attribute__((ext_vector_type(4)));
typedef int          i32x8 __attribute__((ext_vector_type(8)));
typedef int          i32x4 __attribute__((ext_vector_type(4)));

union ABFrag { v16h v; uint4 u[2]; };

#define AS1 __attribute__((address_space(1)))
#define AS3 __attribute__((address_space(3)))

// ---- CDNA5 feature detection (device pass only; host pass takes fallbacks) ----
#if defined(__has_builtin)
#if __has_builtin(__builtin_amdgcn_tensor_load_to_lds) && __has_builtin(__builtin_amdgcn_s_wait_tensorcnt)
#define HAVE_TDM 1
#endif
#if __has_builtin(__builtin_amdgcn_global_load_async_to_lds_b128)
#define HAVE_ASYNC 1
#endif
#if __has_builtin(__builtin_amdgcn_s_wait_asynccnt)
#define HAVE_WAIT_ASYNC 1
#endif
#endif

__device__ __forceinline__ float sigmoidf_(float x) { return 1.0f / (1.0f + expf(-x)); }

// stage 16 bytes global -> LDS; async (ASYNCcnt-tracked) when available
__device__ __forceinline__ void stage16(const _Float16* g, _Float16* l) {
#if defined(HAVE_ASYNC)
    __builtin_amdgcn_global_load_async_to_lds_b128(
        (AS1 v4i*)(uintptr_t)g,
        (AS3 v4i*)(uint32_t)(uintptr_t)l, 0, 0);
#else
    *(uint4*)l = *(const uint4*)g;
#endif
}
__device__ __forceinline__ void wait_async_le1() {
#if defined(HAVE_ASYNC)
#if defined(HAVE_WAIT_ASYNC)
    __builtin_amdgcn_s_wait_asynccnt(1);
#else
    asm volatile("s_wait_asynccnt 0x1" ::: "memory");
#endif
#endif
}
__device__ __forceinline__ void wait_async_le0() {
#if defined(HAVE_ASYNC)
#if defined(HAVE_WAIT_ASYNC)
    __builtin_amdgcn_s_wait_asynccnt(0);
#else
    asm volatile("s_wait_asynccnt 0x0" ::: "memory");
#endif
#endif
}

// ---------------------------------------------------------------------------
// fp32 -> f16 weight conversion (one pass; weights then live in L2 as f16)
// ---------------------------------------------------------------------------
__global__ void k_f32_to_f16(const float* __restrict__ src, _Float16* __restrict__ dst, int n) {
    int i = blockIdx.x * blockDim.x + threadIdx.x;
    if (i < n) dst[i] = (_Float16)src[i];
}

// ---------------------------------------------------------------------------
// Spatial mean pool: h[B,C,T,7,7] -> x16[T,B,C]; also seeds the x-half of the
// attended-concat buffer for both tasks.
// ---------------------------------------------------------------------------
__global__ void k_pool(const float* __restrict__ h, _Float16* __restrict__ x16,
                       _Float16* __restrict__ cat16) {
    int idx = blockIdx.x * blockDim.x + threadIdx.x;   // B*C*T
    if (idx >= BB * CC * TT) return;
    int t = idx % TT;
    int c = (idx / TT) % CC;
    int b = idx / (TT * CC);
    const float* p = h + ((size_t)(b * CC + c) * TT + t) * 49;
    float s = 0.f;
#pragma unroll
    for (int i = 0; i < 49; ++i) s += p[i];
    _Float16 mh = (_Float16)(s * (1.0f / 49.0f));
    int row = t * BB + b;
    x16[(size_t)row * CC + c] = mh;
    cat16[(size_t)row * 2048 + c] = mh;                                  // task 0
    cat16[(size_t)1024 * 2048 + (size_t)row * 2048 + c] = mh;            // task 1
}

// ---------------------------------------------------------------------------
// Generic f16 WMMA GEMM: C[M,N] = A[M,K] @ W[N,K]^T + bias1 + bias2
// Block = 256 (8 waves) covering 16 rows x 128 cols. The whole 16xK A panel
// is DMA'd into LDS once via the Tensor Data Mover (wave 0 issues, TENSORcnt
// waits), then the K loop is barrier-free with software-pipelined B loads.
// Dynamic LDS = 16*K*2 bytes (32KB for K=1024, 64KB for K=2048).
// ---------------------------------------------------------------------------
__global__ void k_gemm_f16(const _Float16* __restrict__ A, int lda,
                           const _Float16* __restrict__ W, int ldw,
                           const float* __restrict__ bias1,
                           const float* __restrict__ bias2,
                           float* __restrict__ Cout, int ldc, int K) {
    extern __shared__ __align__(16) _Float16 Apanel[];
    int tid  = threadIdx.x;
    int wave = tid >> 5, lane = tid & 31;
    int half = lane >> 4, r = lane & 15;
    int m0 = blockIdx.y * 16;
    int n0 = blockIdx.x * 128 + wave * 16;

#if defined(HAVE_TDM)
    if (wave == 0) {
        // Tensor DMA descriptor (D#): 2D tile, 16 rows x K cols of 2-byte elems
        uint64_t ga = (uint64_t)(uintptr_t)(A + (size_t)m0 * lda);
        u32x4 g0;
        g0[0] = 1u;                                            // count=1 (valid)
        g0[1] = (uint32_t)(uintptr_t)Apanel;                   // lds_addr (bytes)
        g0[2] = (uint32_t)ga;                                  // global_addr[31:0]
        g0[3] = ((uint32_t)(ga >> 32) & 0x01FFFFFFu) | (2u << 30); // addr[56:32] | type=2
        i32x8 g1;
        g1[0] = (int)(1u << 16);                               // data_size=1 (2B)
        g1[1] = (int)(((uint32_t)K & 0xFFFFu) << 16);          // tensor_dim0[15:0]
        g1[2] = (int)((((uint32_t)K >> 16) & 0xFFFFu) | (16u << 16)); // dim0 hi | tensor_dim1=16
        g1[3] = (int)(((uint32_t)K & 0xFFFFu) << 16);          // tile_dim0 = K
        g1[4] = 16;                                            // tile_dim1=16, tile_dim2=0
        g1[5] = (int)(uint32_t)lda;                            // tensor_dim0_stride lo32
        g1[6] = 0;
        g1[7] = 0;
        i32x4 z4 = {0, 0, 0, 0};
        i32x8 z8 = {0, 0, 0, 0, 0, 0, 0, 0};
        __builtin_amdgcn_tensor_load_to_lds(g0, g1, z4, z4, z8, 0);
        __builtin_amdgcn_s_wait_tensorcnt(0);
    }
#else
    for (int idx = tid * 8; idx < 16 * K; idx += 256 * 8)
        *(uint4*)&Apanel[idx] = *(const uint4*)&A[(size_t)(idx / K) * lda + (idx % K)];
#endif
    __syncthreads();

    v8f acc = {};
    const _Float16* wrow = &W[(size_t)(n0 + r) * ldw + half * 16];
    ABFrag bcur, bnxt;
    bcur.u[0] = *(const uint4*)wrow;
    bcur.u[1] = *(const uint4*)(wrow + 8);
    for (int kc = 0; kc < K; kc += 32) {
        if (kc + 32 < K) {                       // pipeline next B fragment
            const _Float16* wn = wrow + kc + 32;
            __builtin_prefetch(wn + 32, 0, 3);   // next-next chunk into near cache
            bnxt.u[0] = *(const uint4*)wn;
            bnxt.u[1] = *(const uint4*)(wn + 8);
        }
        ABFrag a;                                 // A frag from resident LDS panel
        a.u[0] = *(const uint4*)&Apanel[(size_t)r * K + kc + half * 16];
        a.u[1] = *(const uint4*)&Apanel[(size_t)r * K + kc + half * 16 + 8];
        acc = __builtin_amdgcn_wmma_f32_16x16x32_f16(false, a.v, false, bcur.v,
                                                     (short)0, acc, false, false);
        bcur = bnxt;
    }
    int n = n0 + r;
    float bv = 0.f;
    if (bias1) bv += bias1[n];
    if (bias2) bv += bias2[n];
#pragma unroll
    for (int q = 0; q < 8; ++q) {
        int row = m0 + q + half * 8;
        Cout[(size_t)row * ldc + n] = acc[q] + bv;
    }
}

// ---------------------------------------------------------------------------
// Zero LSTM state (h double-buffer f16, h fp32, c fp32), 4 instances
// ---------------------------------------------------------------------------
__global__ void k_init_state(_Float16* hA, _Float16* hB, float* h32, float* c) {
    int i = blockIdx.x * blockDim.x + threadIdx.x;
    if (i < 4 * BB * HHID) {
        hA[i] = (_Float16)0.f; hB[i] = (_Float16)0.f;
        h32[i] = 0.f; c[i] = 0.f;
    }
}

// ---------------------------------------------------------------------------
// One LSTM time step for all 4 instances (enc0, enc1, dec0, dec1).
// grid = (H/16 column groups, 4 instances); block = 256 (8 waves).
// h_prev tiles stream into a double-buffered LDS stage via ASYNC global->LDS
// copies (ASYNCcnt), B fragments are software-pipelined, gates fused via an
// LDS z-buffer.
// ---------------------------------------------------------------------------
__global__ void k_lstm_step(const _Float16* __restrict__ hprev,  // [4][B][H] f16
                            _Float16* __restrict__ hnext,        // [4][B][H] f16
                            const _Float16* __restrict__ eWhh16, // [2][4H][H] f16
                            const _Float16* __restrict__ dWhh16, // [2][4H][H] f16
                            const float* __restrict__ Zx,        // [4][T*B][4H]
                            float* __restrict__ cst,             // [4][B][H]
                            float* __restrict__ h32,             // [4][B][H]
                            float* __restrict__ enc_hs,          // [2][T][B][H]
                            int t) {
    __shared__ __align__(16) _Float16 As[2][64 * 32];
    __shared__ float zb[4][64][16];
    int inst = blockIdx.y;
    int n0   = blockIdx.x * 16;
    int tid  = threadIdx.x;
    int wave = tid >> 5, lane = tid & 31, half = lane >> 4, r = lane & 15;

    const _Float16* Abase = hprev + (size_t)inst * BB * HHID;
    const _Float16* W = (inst < 2) ? (eWhh16 + (size_t)inst * G4 * HHID)
                                   : (dWhh16 + (size_t)(inst - 2) * G4 * HHID);
    int id0 = wave * 2;
    int m0t = (id0 >> 2) * 16;
    int g0  = id0 & 3;
    v8f acc0 = {}, acc1 = {};

    int srow = (tid * 8) >> 5, scol = (tid * 8) & 31;     // this thread's stage slot
    // prologue: async-stage k-chunk 0 into buffer 0
    stage16(&Abase[(size_t)srow * HHID + scol], &As[0][tid * 8]);

    const _Float16* w0row = &W[(size_t)(g0 * HHID + n0 + r) * HHID + half * 16];
    const _Float16* w1row = w0row + (size_t)HHID * HHID;  // gate g0+1
    ABFrag b0c, b1c, b0n, b1n;
    b0c.u[0] = *(const uint4*)w0row;        b0c.u[1] = *(const uint4*)(w0row + 8);
    b1c.u[0] = *(const uint4*)w1row;        b1c.u[1] = *(const uint4*)(w1row + 8);

    int cur = 0;
    for (int kc = 0; kc < HHID; kc += 32) {
        bool has = (kc + 32) < HHID;
        if (has) {                              // stage next A chunk + next B frags
            stage16(&Abase[(size_t)srow * HHID + kc + 32 + scol], &As[cur ^ 1][tid * 8]);
            const _Float16* w0n = w0row + kc + 32;
            const _Float16* w1n = w1row + kc + 32;
            b0n.u[0] = *(const uint4*)w0n;  b0n.u[1] = *(const uint4*)(w0n + 8);
            b1n.u[0] = *(const uint4*)w1n;  b1n.u[1] = *(const uint4*)(w1n + 8);
        }
        if (has) wait_async_le1(); else wait_async_le0();   // my copy to As[cur] done
        __syncthreads();                                    // everyone's copy done
        ABFrag a;
        a.u[0] = *(const uint4*)&As[cur][(m0t + r) * 32 + half * 16];
        a.u[1] = *(const uint4*)&As[cur][(m0t + r) * 32 + half * 16 + 8];
        acc0 = __builtin_amdgcn_wmma_f32_16x16x32_f16(false, a.v, false, b0c.v,
                                                      (short)0, acc0, false, false);
        acc1 = __builtin_amdgcn_wmma_f32_16x16x32_f16(false, a.v, false, b1c.v,
                                                      (short)0, acc1, false, false);
        __syncthreads();                        // readers done before buffer reuse
        b0c = b0n; b1c = b1n;
        cur ^= 1;
    }
#pragma unroll
    for (int q = 0; q < 8; ++q) {
        zb[g0][m0t + q + half * 8][r]     = acc0[q];
        zb[g0 + 1][m0t + q + half * 8][r] = acc1[q];
    }
    __syncthreads();

    // fused gate math: z = z_rec + z_x ; PyTorch gate order (i,f,g,o)
    const float* zx = Zx + (size_t)inst * 1024 * G4 + (size_t)t * BB * G4;
    for (int e = tid; e < 64 * 16; e += 256) {
        int m = e >> 4; int col = e & 15; int n = n0 + col;
        float zi = zb[0][m][col] + zx[(size_t)m * G4 + 0 * HHID + n];
        float zf = zb[1][m][col] + zx[(size_t)m * G4 + 1 * HHID + n];
        float zg = zb[2][m][col] + zx[(size_t)m * G4 + 2 * HHID + n];
        float zo = zb[3][m][col] + zx[(size_t)m * G4 + 3 * HHID + n];
        size_t si = (size_t)inst * BB * HHID + (size_t)m * HHID + n;
        float cn = sigmoidf_(zf) * cst[si] + sigmoidf_(zi) * tanhf(zg);
        float hn = sigmoidf_(zo) * tanhf(cn);
        cst[si]   = cn;
        hnext[si] = (_Float16)hn;
        h32[si]   = hn;
        if (inst < 2)
            enc_hs[((size_t)(inst * TT + t) * BB + m) * HHID + n] = hn;
    }
}

// ---------------------------------------------------------------------------
// Attention scores + softmax: P[task][t*B+b][16]
// ---------------------------------------------------------------------------
__global__ void k_attn_scores(const _Float16* __restrict__ x16,
                              const float* __restrict__ enc_hs,
                              const float* __restrict__ aW,   // [NT,16,2048]
                              const float* __restrict__ ab,   // [NT,16]
                              float* __restrict__ P) {
    __shared__ float red[256];
    __shared__ float sc[16];
    int row = blockIdx.x;            // t*B + b
    int task = blockIdx.y;
    int t = row >> 6, b = row & 63;
    int tid = threadIdx.x;
    int s = tid >> 4, part = tid & 15;
    const float* wrow = aW + ((size_t)task * TT + s) * 2048;
    float acc = 0.f;
    for (int i = 0; i < 128; ++i) {
        int k = part * 128 + i;
        float v = (k < 1024) ? (float)x16[(size_t)row * CC + k]
                             : enc_hs[(((size_t)task * TT + t) * BB + b) * HHID + (k - 1024)];
        acc += v * wrow[k];
    }
    red[tid] = acc;
    __syncthreads();
    if (part == 0) {
        float sum = 0.f;
        for (int j = 0; j < 16; ++j) sum += red[s * 16 + j];
        sc[s] = sum + ab[task * TT + s];
    }
    __syncthreads();
    if (tid == 0) {
        float mx = sc[0];
        for (int j = 1; j < 16; ++j) mx = fmaxf(mx, sc[j]);
        float e[16]; float den = 0.f;
        for (int j = 0; j < 16; ++j) { e[j] = expf(sc[j] - mx); den += e[j]; }
        for (int j = 0; j < 16; ++j)
            P[((size_t)task * 1024 + row) * 16 + j] = e[j] / den;
    }
}

// ---------------------------------------------------------------------------
// applied = einsum("tbs,sbh->tbh"); result into the enc-half of cat16 (f16)
// ---------------------------------------------------------------------------
__global__ void k_attn_apply(const float* __restrict__ P,
                             const float* __restrict__ enc_hs,
                             _Float16* __restrict__ cat16) {
    size_t idx = (size_t)blockIdx.x * blockDim.x + threadIdx.x;
    if (idx >= (size_t)NTASK * TT * BB * HHID) return;
    int hcol = idx % HHID;
    int b = (idx / HHID) % BB;
    int t = (idx / ((size_t)HHID * BB)) % TT;
    int task = idx / ((size_t)HHID * BB * TT);
    const float* p = P + ((size_t)task * 1024 + t * BB + b) * 16;
    float acc = 0.f;
#pragma unroll
    for (int s = 0; s < 16; ++s)
        acc += p[s] * enc_hs[(((size_t)task * TT + s) * BB + b) * HHID + hcol];
    cat16[(size_t)task * 1024 * 2048 + (size_t)(t * BB + b) * 2048 + 1024 + hcol] = (_Float16)acc;
}

// ---------------------------------------------------------------------------
// FC heads: logits0 [64,125], logits1 [64,352] from concat(dec0_h, dec1_h)
// ---------------------------------------------------------------------------
__global__ void k_fc(const float* __restrict__ h32,   // [4][B][H]; dec = inst 2,3
                     const float* __restrict__ w0, const float* __restrict__ b0v,
                     const float* __restrict__ w1, const float* __restrict__ b1v,
                     float* __restrict__ out) {
    int idx = blockIdx.x * blockDim.x + threadIdx.x;
    if (idx >= BB * (NC0 + NC1)) return;
    int b = idx / (NC0 + NC1);
    int j = idx % (NC0 + NC1);
    const float* wrow; float bias; float* dst;
    if (j < NC0) { wrow = w0 + (size_t)j * 2048; bias = b0v[j]; dst = out + (size_t)b * NC0 + j; }
    else { int jj = j - NC0; wrow = w1 + (size_t)jj * 2048; bias = b1v[jj];
           dst = out + (size_t)BB * NC0 + (size_t)b * NC1 + jj; }
    const float* hd0 = h32 + (size_t)2 * BB * HHID + (size_t)b * HHID;
    const float* hd1 = h32 + (size_t)3 * BB * HHID + (size_t)b * HHID;
    float acc = bias;
    for (int k = 0; k < 1024; ++k) acc += hd0[k] * wrow[k];
    for (int k = 0; k < 1024; ++k) acc += hd1[k] * wrow[k + 1024];
    *dst = acc;
}

// ---------------------------------------------------------------------------
extern "C" void kernel_launch(void* const* d_in, const int* in_sizes, int n_in,
                              void* d_out, int out_size, void* d_ws, size_t ws_size,
                              hipStream_t stream) {
    (void)in_sizes; (void)n_in; (void)out_size; (void)ws_size;
    const float* h        = (const float*)d_in[0];
    const float* enc_Wih  = (const float*)d_in[1];
    const float* enc_Whh  = (const float*)d_in[2];
    const float* enc_bih  = (const float*)d_in[3];
    const float* enc_bhh  = (const float*)d_in[4];
    const float* dec_Wih  = (const float*)d_in[5];
    const float* dec_Whh  = (const float*)d_in[6];
    const float* dec_bih  = (const float*)d_in[7];
    const float* dec_bhh  = (const float*)d_in[8];
    const float* attn_W   = (const float*)d_in[9];
    const float* attn_b   = (const float*)d_in[10];
    const float* comb_W   = (const float*)d_in[11];
    const float* comb_b   = (const float*)d_in[12];
    const float* fc0_W    = (const float*)d_in[13];
    const float* fc0_b    = (const float*)d_in[14];
    const float* fc1_W    = (const float*)d_in[15];
    const float* fc1_b    = (const float*)d_in[16];
    float* out = (float*)d_out;

    // --- workspace carve-out ---
    char* ws = (char*)d_ws;
    size_t off = 0;
    auto carve = [&](size_t bytes) { void* p = ws + off; off = (off + bytes + 255) & ~(size_t)255; return p; };
    const size_t WIH_E = (size_t)NTASK * G4 * HHID;        // 8,388,608 elems
    _Float16* x16     = (_Float16*)carve((size_t)TT * BB * CC * 2);
    _Float16* eWih16  = (_Float16*)carve(WIH_E * 2);
    _Float16* eWhh16  = (_Float16*)carve(WIH_E * 2);
    _Float16* dWih16  = (_Float16*)carve(WIH_E * 2);
    _Float16* dWhh16  = (_Float16*)carve(WIH_E * 2);
    _Float16* combW16 = (_Float16*)carve((size_t)NTASK * HHID * 2048 * 2);
    _Float16* cat16   = (_Float16*)carve((size_t)NTASK * 1024 * 2048 * 2);
    float*    Zx      = (float*)carve((size_t)4 * 1024 * G4 * 4);   // 64MB
    _Float16* hbufA   = (_Float16*)carve((size_t)4 * BB * HHID * 2);
    _Float16* hbufB   = (_Float16*)carve((size_t)4 * BB * HHID * 2);
    float*    h32     = (float*)carve((size_t)4 * BB * HHID * 4);
    float*    cst     = (float*)carve((size_t)4 * BB * HHID * 4);
    float*    enc_hs  = (float*)carve((size_t)NTASK * TT * BB * HHID * 4);
    float*    attnP   = (float*)carve((size_t)NTASK * 1024 * 16 * 4);

    // --- 1. convert weights to f16 once ---
    {
        int n = (int)WIH_E;
        k_f32_to_f16<<<(n + 255) / 256, 256, 0, stream>>>(enc_Wih, eWih16, n);
        k_f32_to_f16<<<(n + 255) / 256, 256, 0, stream>>>(enc_Whh, eWhh16, n);
        k_f32_to_f16<<<(n + 255) / 256, 256, 0, stream>>>(dec_Wih, dWih16, n);
        k_f32_to_f16<<<(n + 255) / 256, 256, 0, stream>>>(dec_Whh, dWhh16, n);
        int nc = NTASK * HHID * 2048;
        k_f32_to_f16<<<(nc + 255) / 256, 256, 0, stream>>>(comb_W, combW16, nc);
    }

    // --- 2. spatial mean pool (205MB streamed once) ---
    k_pool<<<(BB * CC * TT) / 256, 256, 0, stream>>>(h, x16, cat16);

    // --- 3. hoisted input GEMMs: Zx[inst] = x @ Wih^T + bih + bhh (WMMA+TDM) ---
    {
        dim3 grid(G4 / 128, (TT * BB) / 16);
        size_t shmem = (size_t)16 * CC * 2;    // 32KB A panel
        for (int task = 0; task < NTASK; ++task) {
            k_gemm_f16<<<grid, 256, shmem, stream>>>(x16, CC,
                eWih16 + (size_t)task * G4 * HHID, CC,
                enc_bih + task * G4, enc_bhh + task * G4,
                Zx + (size_t)task * 1024 * G4, G4, CC);
            k_gemm_f16<<<grid, 256, shmem, stream>>>(x16, CC,
                dWih16 + (size_t)task * G4 * HHID, CC,
                dec_bih + task * G4, dec_bhh + task * G4,
                Zx + (size_t)(2 + task) * 1024 * G4, G4, CC);
        }
    }

    // --- 4. recurrent loop: 16 sequential steps, 4 LSTMs fused per step ---
    k_init_state<<<(4 * BB * HHID) / 256, 256, 0, stream>>>(hbufA, hbufB, h32, cst);
    {
        dim3 grid(HHID / 16, 4);
        for (int t = 0; t < TT; ++t) {
            const _Float16* hp = (t & 1) ? hbufB : hbufA;
            _Float16*       hn = (t & 1) ? hbufA : hbufB;
            k_lstm_step<<<grid, 256, 0, stream>>>(hp, hn, eWhh16, dWhh16, Zx,
                                                  cst, h32, enc_hs, t);
        }
    }

    // --- 5. attention: scores+softmax, then applied into cat buffer ---
    {
        dim3 g1(TT * BB, NTASK);
        k_attn_scores<<<g1, 256, 0, stream>>>(x16, enc_hs, attn_W, attn_b, attnP);
        size_t tot = (size_t)NTASK * TT * BB * HHID;
        k_attn_apply<<<(unsigned)(tot / 256), 256, 0, stream>>>(attnP, enc_hs, cat16);
    }

    // --- 6. attended = cat @ comb_W^T + comb_b (WMMA+TDM), into d_out ---
    {
        dim3 grid(HHID / 128, (TT * BB) / 16);
        size_t shmem = (size_t)16 * 2048 * 2;  // 64KB A panel
        size_t attended_off = (size_t)BB * NC0 + (size_t)BB * NC1;   // after logits
        for (int task = 0; task < NTASK; ++task) {
            k_gemm_f16<<<grid, 256, shmem, stream>>>(
                cat16 + (size_t)task * 1024 * 2048, 2048,
                combW16 + (size_t)task * HHID * 2048, 2048,
                comb_b + task * HHID, nullptr,
                out + attended_off + (size_t)task * TT * BB * HHID, HHID, 2048);
        }
    }

    // --- 7. FC heads ---
    {
        int tot = BB * (NC0 + NC1);
        k_fc<<<(tot + 255) / 256, 256, 0, stream>>>(h32, fc0_W, fc0_b, fc1_W, fc1_b, out);
    }
}
